// SVNSDENoVolatilityChannel_5935644803472
// MI455X (gfx1250) — matrix operand order repaired and checked by reading, not compiled, not run on hardware
//
#include <hip/hip_runtime.h>
#include <hip/hip_bf16.h>
#include <math.h>

// ---------------------------------------------------------------------------
// SVNSDE loss on MI455X (gfx1250, wave32).
// All heavy GEMMs via v_wmma_f32_16x16x32_f16 (f16 inputs, f32 accum).
// Weights pre-transposed to f16 [n][k] in ws -> each B fragment is one
// contiguous 32B load, fully L2-resident. Intensity kernel processes a
// 32-row pair per wave so every B fragment feeds TWO wmmas. z_traj read
// exactly once (KL sum fused into A-loads) + global_prefetch for the next
// grid-stride tile. Deterministic two-stage reduction (no FP atomics).
// ---------------------------------------------------------------------------

typedef __attribute__((ext_vector_type(16))) _Float16 v16h;
typedef __attribute__((ext_vector_type(8)))  float    v8f;

#define D_LAT  128
#define D_HID  512
#define E_CNT  1024
#define S_CNT  64
#define NSTEPS 8192

__device__ __forceinline__ float softplusf(float x) {
    return (x > 20.f) ? x : log1pf(expf(x));
}

// A fragment (16x32 f16) from f32 row-major matrix, contiguous k order (same as B):
// lane<16: M=lane, K in {0..15}; lane>=16: M=lane-16, K in {16..31}
__device__ __forceinline__ v16h load_a_f32(const float* __restrict__ tile_base,
                                           int ld, int kbase, int lane,
                                           float* sumsq = nullptr) {
    const int m  = lane & 15;
    const int hi = lane >> 4;
    const float* p = tile_base + (size_t)m * ld;
    v16h a;
#pragma unroll
    for (int v = 0; v < 8; ++v) {
        const int k = kbase + (hi << 4) + (v << 1);
        const float2 f = *reinterpret_cast<const float2*>(p + k);
        if (sumsq) *sumsq += f.x * f.x + f.y * f.y;
        a[2 * v]     = (_Float16)f.x;
        a[2 * v + 1] = (_Float16)f.y;
    }
    return a;
}

// A fragment from f16 row-major LDS buffer (same layout as above).
__device__ __forceinline__ v16h load_a_lds(const _Float16* hb, int ldh,
                                           int kbase, int lane) {
    const int m  = lane & 15;
    const int hi = lane >> 4;
    const _Float16* p = hb + m * ldh;
    v16h a;
#pragma unroll
    for (int v = 0; v < 8; ++v) {
        const int k = kbase + (hi << 4) + (v << 1);
        a[2 * v]     = p[k];
        a[2 * v + 1] = p[k + 1];
    }
    return a;
}

// B fragment (32x16 f16) from transposed f16 weight Wt[n][k] (ldk = K dim).
// lane<16: N=lane, K=kbase..kbase+15; lane>=16: N=lane-16, K=kbase+16..+31.
// 16 contiguous halves -> one 32-byte load (2x global_load_b128).
__device__ __forceinline__ v16h load_b_t16(const _Float16* __restrict__ Wt,
                                           int ldk, int ncol_base, int kbase,
                                           int lane) {
    const int n  = ncol_base + (lane & 15);
    const int hi = lane >> 4;
    const _Float16* p = Wt + (size_t)n * ldk + kbase + (hi << 4);
    return *reinterpret_cast<const v16h*>(p);
}

__device__ __forceinline__ v8f wmma16(v16h a, v16h b, v8f c) {
    return __builtin_amdgcn_wmma_f32_16x16x32_f16(false, a, false, b,
                                                  (short)0, c, false, false);
}

// ---------------------------------------------------------------------------
// prep: transpose+convert weights to f16 [n][k] layout in ws.
// ---------------------------------------------------------------------------
__global__ __launch_bounds__(256) void prep_kernel(
    const float* __restrict__ W1, const float* __restrict__ Wd1,
    const float* __restrict__ Wd2, const float* __restrict__ Wp,
    _Float16* __restrict__ W1t, _Float16* __restrict__ Wd1t,
    _Float16* __restrict__ Wd2t, _Float16* __restrict__ Wpt) {
    const int idx    = blockIdx.x * blockDim.x + threadIdx.x;
    const int stride = gridDim.x * blockDim.x;
    for (int i = idx; i < D_LAT * D_HID; i += stride) {   // W1/Wd1: [k=128][n=512]
        const int k = i / D_HID, n = i % D_HID;
        W1t [n * D_LAT + k] = (_Float16)W1[i];
        Wd1t[n * D_LAT + k] = (_Float16)Wd1[i];
    }
    for (int i = idx; i < D_HID * D_LAT; i += stride) {   // Wd2: [k=512][n=128]
        const int k = i / D_LAT, n = i % D_LAT;
        Wd2t[n * D_HID + k] = (_Float16)Wd2[i];
    }
    for (int i = idx; i < D_LAT * D_LAT; i += stride) {   // Wp: [k=128][n=128]
        const int k = i / D_LAT, n = i % D_LAT;
        Wpt[n * D_LAT + k] = (_Float16)Wp[i];
    }
}

// ---------------------------------------------------------------------------
// marks_proj = event_marks @ Wp + bp  (1024x128) — one wave per 16x16 tile.
// ---------------------------------------------------------------------------
__global__ __launch_bounds__(128) void marks_kernel(
    const float* __restrict__ EM, const _Float16* __restrict__ Wpt,
    const float* __restrict__ bp, float* __restrict__ MP) {
    const int lane = threadIdx.x & 31;
    const int wave = (blockIdx.x * blockDim.x + threadIdx.x) >> 5;  // 0..511
    const int rt = wave >> 3, nt = wave & 7;
    const float* erow = EM + (size_t)rt * 16 * D_LAT;

    const v16h a0 = load_a_f32(erow, D_LAT, 0,  lane);
    const v16h a1 = load_a_f32(erow, D_LAT, 32, lane);
    const v16h a2 = load_a_f32(erow, D_LAT, 64, lane);
    const v16h a3 = load_a_f32(erow, D_LAT, 96, lane);

    const int n  = nt * 16 + (lane & 15);
    const float bn = bp[n];
    v8f c = {bn, bn, bn, bn, bn, bn, bn, bn};
    c = wmma16(a0, load_b_t16(Wpt, D_LAT, nt * 16, 0,  lane), c);
    c = wmma16(a1, load_b_t16(Wpt, D_LAT, nt * 16, 32, lane), c);
    c = wmma16(a2, load_b_t16(Wpt, D_LAT, nt * 16, 64, lane), c);
    c = wmma16(a3, load_b_t16(Wpt, D_LAT, nt * 16, 96, lane), c);

    const int m0 = (lane >> 4) * 8;
#pragma unroll
    for (int v = 0; v < 8; ++v)
        MP[(size_t)(rt * 16 + m0 + v) * D_LAT + n] = c[v];
}

// ---------------------------------------------------------------------------
// intensity: lam = softplus(relu(z@W1+b1)@W2 + b2 + base)
// 32-row pair per wave: one B fragment feeds two wmmas.
// mode 0 (events): partial = sum log(lam + 1e-8)
// mode 1 (traj):   partial = sum lam*dt ; partial_sq = sum z^2 (fused KL)
// ---------------------------------------------------------------------------
__global__ __launch_bounds__(256) void intensity_kernel(
    const float* __restrict__ Z, int nrows,
    const _Float16* __restrict__ W1t, const float* __restrict__ b1,
    const float* __restrict__ W2, const float* __restrict__ b2,
    const float* __restrict__ base, const float* __restrict__ times,
    float* __restrict__ partial, float* __restrict__ partial_sq, int mode) {
    const int lane   = threadIdx.x & 31;
    const int wave   = (blockIdx.x * blockDim.x + threadIdx.x) >> 5;
    const int nwaves = (gridDim.x * blockDim.x) >> 5;
    const int npair  = nrows >> 5;                 // 32-row pairs
    const float bias = b2[0] + base[0];

    float local = 0.f, localsq = 0.f;
    for (int pair = wave; pair < npair; pair += nwaves) {
        const float* z0 = Z + (size_t)pair * 32 * D_LAT;
        const float* z1 = z0 + 16 * D_LAT;

        // prefetch next grid-stride pair (streams from HBM)
        if (pair + nwaves < npair) {
            const char* nx = (const char*)(Z + (size_t)(pair + nwaves) * 32 * D_LAT)
                             + lane * 512;        // one 512B row per lane
            __builtin_prefetch(nx, 0, 1);
            __builtin_prefetch(nx + 128, 0, 1);
            __builtin_prefetch(nx + 256, 0, 1);
            __builtin_prefetch(nx + 384, 0, 1);
        }

        float sq = 0.f;
        v16h a0[4], a1[4];
#pragma unroll
        for (int kc = 0; kc < 4; ++kc) {
            a0[kc] = load_a_f32(z0, D_LAT, kc * 32, lane, &sq);
            a1[kc] = load_a_f32(z1, D_LAT, kc * 32, lane, &sq);
        }

        float sacc0[8] = {0.f, 0.f, 0.f, 0.f, 0.f, 0.f, 0.f, 0.f};
        float sacc1[8] = {0.f, 0.f, 0.f, 0.f, 0.f, 0.f, 0.f, 0.f};
        for (int nt = 0; nt < D_HID / 16; ++nt) {           // 32 N-tiles
            const int n  = nt * 16 + (lane & 15);
            const float bn = b1[n];
            v8f c0 = {bn, bn, bn, bn, bn, bn, bn, bn};
            v8f c1 = c0;
#pragma unroll
            for (int kc = 0; kc < 4; ++kc) {
                const v16h b = load_b_t16(W1t, D_LAT, nt * 16, kc * 32, lane);
                c0 = wmma16(a0[kc], b, c0);                  // B reused twice
                c1 = wmma16(a1[kc], b, c1);
            }
            const float w2 = W2[n];
#pragma unroll
            for (int v = 0; v < 8; ++v) {
                sacc0[v] += fmaxf(c0[v], 0.f) * w2;          // h @ W2 partials
                sacc1[v] += fmaxf(c1[v], 0.f) * w2;
            }
        }
        // reduce over the 16 lanes of each half (same rows)
#pragma unroll
        for (int msk = 8; msk; msk >>= 1)
#pragma unroll
            for (int v = 0; v < 8; ++v) {
                sacc0[v] += __shfl_xor(sacc0[v], msk, 32);
                sacc1[v] += __shfl_xor(sacc1[v], msk, 32);
            }

        if ((lane & 15) == 0) {
            if (mode == 0) {
#pragma unroll
                for (int v = 0; v < 8; ++v) {
                    local += logf(softplusf(sacc0[v] + bias) + 1e-8f);
                    local += logf(softplusf(sacc1[v] + bias) + 1e-8f);
                }
            } else {
                const int t = pair >> 1;                     // 2 pairs per step
                const float dt = times[t + 1] - times[t];
#pragma unroll
                for (int v = 0; v < 8; ++v) {
                    local += softplusf(sacc0[v] + bias) * dt;
                    local += softplusf(sacc1[v] + bias) * dt;
                }
            }
        }
        localsq += sq;
    }
    // full wave reduction, one deterministic partial per wave
#pragma unroll
    for (int msk = 16; msk; msk >>= 1) {
        local   += __shfl_xor(local, msk, 32);
        localsq += __shfl_xor(localsq, msk, 32);
    }
    if (lane == 0) {
        partial[wave] = local;
        if (partial_sq) partial_sq[wave] = localsq;
    }
}

// ---------------------------------------------------------------------------
// decoder: h = relu(z_ev@Wd1+bd1) -> LDS ; mu = h@Wd2+bd2 ;
// partial = sum over rows of ||marks_proj[e] - mu||^2
// One wave per 16-row tile (all 16 rows share event e). 2 waves/block.
// ---------------------------------------------------------------------------
#define HPAD 520
__global__ __launch_bounds__(64) void decoder_kernel(
    const float* __restrict__ Z,
    const _Float16* __restrict__ Wd1t, const float* __restrict__ bd1,
    const _Float16* __restrict__ Wd2t, const float* __restrict__ bd2,
    const float* __restrict__ MP, float* __restrict__ partial, int nrows) {
    __shared__ _Float16 hbuf[2][16][HPAD];
    const int lane   = threadIdx.x & 31;
    const int wslot  = threadIdx.x >> 5;
    const int wave   = (blockIdx.x * blockDim.x + threadIdx.x) >> 5;
    const int nwaves = (gridDim.x * blockDim.x) >> 5;
    const int ntile  = nrows >> 4;
    _Float16 (*hb)[HPAD] = hbuf[wslot];

    float local = 0.f;
    for (int tile = wave; tile < ntile; tile += nwaves) {
        const float* zrow = Z + (size_t)tile * 16 * D_LAT;

        if (tile + nwaves < ntile) {               // prefetch next z tile (8KB)
            const char* nx = (const char*)(Z + (size_t)(tile + nwaves) * 16 * D_LAT)
                             + lane * 256;
            __builtin_prefetch(nx, 0, 1);
            __builtin_prefetch(nx + 128, 0, 1);
        }

        const v16h a0 = load_a_f32(zrow, D_LAT, 0,  lane);
        const v16h a1 = load_a_f32(zrow, D_LAT, 32, lane);
        const v16h a2 = load_a_f32(zrow, D_LAT, 64, lane);
        const v16h a3 = load_a_f32(zrow, D_LAT, 96, lane);

        // stage h tile (16 x 512 f16) in LDS, own wave only
        const int m0 = (lane >> 4) * 8;
        for (int nt = 0; nt < D_HID / 16; ++nt) {
            const int n  = nt * 16 + (lane & 15);
            const float bn = bd1[n];
            v8f c = {bn, bn, bn, bn, bn, bn, bn, bn};
            c = wmma16(a0, load_b_t16(Wd1t, D_LAT, nt * 16, 0,  lane), c);
            c = wmma16(a1, load_b_t16(Wd1t, D_LAT, nt * 16, 32, lane), c);
            c = wmma16(a2, load_b_t16(Wd1t, D_LAT, nt * 16, 64, lane), c);
            c = wmma16(a3, load_b_t16(Wd1t, D_LAT, nt * 16, 96, lane), c);
#pragma unroll
            for (int v = 0; v < 8; ++v)
                hb[m0 + v][n] = (_Float16)fmaxf(c[v], 0.f);
        }

        const int e = tile >> 2;                // 4 tiles per event (S=64)
        const float* xrow = MP + (size_t)e * D_LAT;
        float ssq[8] = {0.f, 0.f, 0.f, 0.f, 0.f, 0.f, 0.f, 0.f};
        for (int mt = 0; mt < D_LAT / 16; ++mt) {            // 8 mu N-tiles
            const int n  = mt * 16 + (lane & 15);
            const float bn = bd2[n];
            v8f c = {bn, bn, bn, bn, bn, bn, bn, bn};
            for (int kc = 0; kc < D_HID / 32; ++kc) {        // 16 K-chunks
                const v16h ah = load_a_lds(&hb[0][0], HPAD, kc * 32, lane);
                const v16h bh = load_b_t16(Wd2t, D_HID, mt * 16, kc * 32, lane);
                c = wmma16(ah, bh, c);
            }
            const float x = xrow[n];
#pragma unroll
            for (int v = 0; v < 8; ++v) {
                const float d = x - c[v];
                ssq[v] += d * d;
            }
        }
#pragma unroll
        for (int msk = 8; msk; msk >>= 1)
#pragma unroll
            for (int v = 0; v < 8; ++v)
                ssq[v] += __shfl_xor(ssq[v], msk, 32);
        if ((lane & 15) == 0) {
#pragma unroll
            for (int v = 0; v < 8; ++v) local += ssq[v];
        }
    }
#pragma unroll
    for (int msk = 16; msk; msk >>= 1) local += __shfl_xor(local, msk, 32);
    if (lane == 0) partial[wave] = local;
}

// ---------------------------------------------------------------------------
// finalize: deterministic tree reduction of per-wave partials -> scalar loss
// ---------------------------------------------------------------------------
__device__ float block_sum(const float* __restrict__ p, int n, float* sbuf) {
    float s = 0.f;
    for (int i = threadIdx.x; i < n; i += blockDim.x) s += p[i];
    sbuf[threadIdx.x] = s;
    __syncthreads();
    for (int off = blockDim.x >> 1; off > 0; off >>= 1) {
        if ((int)threadIdx.x < off) sbuf[threadIdx.x] += sbuf[threadIdx.x + off];
        __syncthreads();
    }
    const float r = sbuf[0];
    __syncthreads();
    return r;
}

__global__ __launch_bounds__(256) void finalize_kernel(
    const float* __restrict__ p_ev, const float* __restrict__ p_tr,
    const float* __restrict__ p_sq, const float* __restrict__ p_rec,
    float* __restrict__ out) {
    __shared__ float sbuf[256];
    const float a_ev  = block_sum(p_ev,  2048, sbuf);   // sum log(lam_ev+1e-8)
    const float a_tr  = block_sum(p_tr,  8192, sbuf);   // sum lam_traj*dt
    const float a_sq  = block_sum(p_sq,  8192, sbuf);   // sum z_traj^2
    const float a_rec = block_sum(p_rec, 4096, sbuf);   // sum ||x-mu||^2
    if (threadIdx.x == 0) {
        const float LOG2PI = 1.8378770664093454f;
        const float log_int  = a_ev / (float)S_CNT;
        const float integral = a_tr / (float)S_CNT;
        const float recon    = -0.5f * a_rec / (float)S_CNT
                               - (float)E_CNT * 0.5f * (float)D_LAT * LOG2PI;
        const float kl = 0.01f * a_sq / (float)((size_t)NSTEPS * S_CNT * D_LAT);
        out[0] = -(log_int - integral + recon - kl);
    }
}

// ---------------------------------------------------------------------------
extern "C" void kernel_launch(void* const* d_in, const int* in_sizes, int n_in,
                              void* d_out, int out_size, void* d_ws, size_t ws_size,
                              hipStream_t stream) {
    const float* event_marks = (const float*)d_in[0];
    const float* z_events    = (const float*)d_in[1];
    const float* z_traj      = (const float*)d_in[2];
    const float* times       = (const float*)d_in[3];
    const float* Wp          = (const float*)d_in[4];
    const float* bp          = (const float*)d_in[5];
    const float* W1          = (const float*)d_in[6];
    const float* b1          = (const float*)d_in[7];
    const float* W2          = (const float*)d_in[8];
    const float* b2          = (const float*)d_in[9];
    const float* base        = (const float*)d_in[10];
    const float* Wd1         = (const float*)d_in[11];
    const float* bd1         = (const float*)d_in[12];
    const float* Wd2         = (const float*)d_in[13];
    const float* bd2         = (const float*)d_in[14];
    float* out = (float*)d_out;

    // ws layout (1 MiB total)
    char* ws = (char*)d_ws;
    float*    marks_proj = (float*)(ws);                       // 1024*128 f32 = 512 KiB
    _Float16* W1t  = (_Float16*)(ws + 524288);                 // 512*128 f16 = 128 KiB
    _Float16* Wd1t = (_Float16*)(ws + 655360);                 // 128 KiB
    _Float16* Wd2t = (_Float16*)(ws + 786432);                 // 128 KiB
    _Float16* Wpt  = (_Float16*)(ws + 917504);                 // 32 KiB
    float* p_ev  = (float*)(ws + 950272);                      // 2048 f
    float* p_tr  = (float*)(ws + 966656);                      // 8192 f
    float* p_sq  = (float*)(ws + 999424);                      // 8192 f
    float* p_rec = (float*)(ws + 1032192);                     // 4096 f

    prep_kernel<<<256, 256, 0, stream>>>(W1, Wd1, Wd2, Wp, W1t, Wd1t, Wd2t, Wpt);
    marks_kernel<<<128, 128, 0, stream>>>(event_marks, Wpt, bp, marks_proj);
    // events: 65536 rows -> 2048 pairs, 2048 waves (256 blk * 8 waves)
    intensity_kernel<<<256, 256, 0, stream>>>(z_events, E_CNT * S_CNT,
        W1t, b1, W2, b2, base, times, p_ev, nullptr, 0);
    // traj: 524288 rows -> 16384 pairs, 8192 waves x 2 pairs
    intensity_kernel<<<1024, 256, 0, stream>>>(z_traj, NSTEPS * S_CNT,
        W1t, b1, W2, b2, base, times, p_tr, p_sq, 1);
    // decoder: 4096 tiles, 4096 waves (2048 blk * 2 waves)
    decoder_kernel<<<2048, 64, 0, stream>>>(z_events, Wd1t, bd1, Wd2t, bd2,
        marks_proj, p_rec, E_CNT * S_CNT);
    finalize_kernel<<<1, 256, 0, stream>>>(p_ev, p_tr, p_sq, p_rec, out);
}
